// Moe_36739150250509
// MI455X (gfx1250) — matrix operand order repaired and checked by reading, not compiled, not run on hardware
//
#include <hip/hip_runtime.h>

// MoE: E=8 experts, top-2, H=1024, F=2816, T=B*S=1024 tokens, fp32 in/out.
// Routed-only compute, bf16 WMMA (v_wmma_f32_16x16x32_bf16) with f32
// accumulate; x-tile and silu(g)*u chunks staged in LDS; deterministic
// slot-partial output (no atomics).

#define NE 8
#define HD 1024
#define FD 2816
#define NT 1024
#define TM 16
#define FCHUNK 704   // FD/4
#define NWAVE 8

typedef __attribute__((ext_vector_type(16))) __bf16 v16bf;
typedef __attribute__((ext_vector_type(8)))  __bf16 v8bf;
typedef __attribute__((ext_vector_type(8)))  float  v8f;

__device__ __forceinline__ __bf16 f2bf(float f) {
  return static_cast<__bf16>(f);   // native bf16 cvt on gfx1250
}

__device__ __forceinline__ unsigned short f2bfu(float f) {
  return __builtin_bit_cast(unsigned short, static_cast<__bf16>(f));
}

__device__ __forceinline__ v16bf cat8(v8bf lo, v8bf hi) {
  return __builtin_shufflevector(lo, hi, 0,1,2,3,4,5,6,7,8,9,10,11,12,13,14,15);
}

// ---------------- Router: softmax + top-2 + renorm, dense weights [T,E] -----
__global__ void moe_router(const float* __restrict__ x,
                           const float* __restrict__ rw,
                           float* __restrict__ wdense) {
  const int t = blockIdx.x * blockDim.x + threadIdx.x;
  if (t >= NT) return;
  float acc[NE];
#pragma unroll
  for (int e = 0; e < NE; ++e) acc[e] = 0.f;
  const float4* xv = (const float4*)(x + (size_t)t * HD);
  for (int h = 0; h < HD / 4; ++h) {
    const float4 xx = xv[h];
#pragma unroll
    for (int e = 0; e < NE; ++e) {
      const float4 ww = ((const float4*)(rw + (size_t)e * HD))[h];
      acc[e] += xx.x * ww.x + xx.y * ww.y + xx.z * ww.z + xx.w * ww.w;
    }
  }
  float mx = acc[0];
#pragma unroll
  for (int e = 1; e < NE; ++e) mx = fmaxf(mx, acc[e]);
  float p[NE]; float s = 0.f;
#pragma unroll
  for (int e = 0; e < NE; ++e) { p[e] = __expf(acc[e] - mx); s += p[e]; }
  const float inv = __builtin_amdgcn_rcpf(s);
#pragma unroll
  for (int e = 0; e < NE; ++e) p[e] *= inv;
  int i1 = 0;
#pragma unroll
  for (int e = 1; e < NE; ++e) if (p[e] > p[i1]) i1 = e;
  int i2 = (i1 == 0) ? 1 : 0;
#pragma unroll
  for (int e = 0; e < NE; ++e) if (e != i1 && p[e] > p[i2]) i2 = e;
  const float norm = __builtin_amdgcn_rcpf(p[i1] + p[i2]);
#pragma unroll
  for (int e = 0; e < NE; ++e)
    wdense[(size_t)t * NE + e] = (e == i1 || e == i2) ? p[e] * norm : 0.f;
}

// ------------- Compaction: one wave per expert, deterministic order --------
__global__ void moe_compact(const float* __restrict__ wdense,
                            int* __restrict__ cnt,
                            int* __restrict__ toks,
                            float* __restrict__ wts,
                            int* __restrict__ slots) {
  const int e = threadIdx.x >> 5;
  const int lane = threadIdx.x & 31;
  int base = 0;
  for (int t0 = 0; t0 < NT; t0 += 32) {
    const int t = t0 + lane;
    const float w = wdense[(size_t)t * NE + e];
    const bool sel = (w > 0.f);
    const unsigned m = (unsigned)__ballot(sel);
    const int off = __popc(m & ((1u << lane) - 1u));
    if (sel) {
      int slot = 0;                         // 0 or 1: rank by expert index
      for (int ep = 0; ep < e; ++ep) slot += (wdense[(size_t)t * NE + ep] > 0.f);
      const int pos = e * NT + base + off;
      toks[pos] = t; wts[pos] = w; slots[pos] = slot;
    }
    base += __popc(m);
  }
  if (lane == 0) cnt[e] = base;
}

// ---------------- Fused routed SwiGLU FFN, bf16 WMMA -----------------------
__global__ __launch_bounds__(256)
void moe_ffn(const float* __restrict__ x,
             const float* __restrict__ gate,
             const float* __restrict__ up,
             const float* __restrict__ down,
             const int* __restrict__ toks,
             const float* __restrict__ wts,
             const int* __restrict__ slots,
             const int* __restrict__ cnt,
             float* __restrict__ part) {
  __shared__ unsigned short xb[TM * HD];       // 32 KB: x tile, bf16
  __shared__ unsigned short mb[TM * FCHUNK];   // 22 KB: silu(g)*u chunk, bf16
  __shared__ int   tok_s[TM];
  __shared__ float w_s[TM];
  __shared__ int   slot_s[TM];

  const int e = blockIdx.y;
  const int tile = blockIdx.x;
  const int n = cnt[e];
  if (tile * TM >= n) return;

  const int tid = threadIdx.x;
  const int lane = tid & 31;
  const int wave = tid >> 5;
  const int Mrow = lane & 15;       // A row / B,C column within a 16-wide tile
  const int khalf = lane >> 4;      // which K half this lane holds

  if (tid < TM) {
    const int idx = tile * TM + tid;
    if (idx < n) {
      tok_s[tid]  = toks[e * NT + idx];
      w_s[tid]    = wts[e * NT + idx];
      slot_s[tid] = slots[e * NT + idx];
    } else { tok_s[tid] = 0; w_s[tid] = 0.f; slot_s[tid] = 0; }
  }
  __syncthreads();

  for (int i = tid; i < TM * HD; i += 256) {   // gather 16 token rows -> LDS bf16
    const int r = i >> 10;
    const int c = i & (HD - 1);
    xb[i] = f2bfu(x[(size_t)tok_s[r] * HD + c]);
  }
  __syncthreads();

  const float* gbase = gate + (size_t)e * HD * FD;
  const float* ubase = up   + (size_t)e * HD * FD;
  const float* dbase = down + (size_t)e * FD * HD;

  const v8f vzero = {};
  v8f acc[8];
#pragma unroll
  for (int j = 0; j < 8; ++j) acc[j] = vzero;   // 16x128 fp32 output per wave

  for (int chunk = 0; chunk < FD / FCHUNK; ++chunk) {
    const int fbase = chunk * FCHUNK;

    // Phase A: waves split F-tiles; m = silu(x@gate) * (x@up) for this chunk
    for (int ft = wave; ft < FCHUNK / 16; ft += NWAVE) {
      const int f0 = ft * 16;
      const int ncol = fbase + f0 + Mrow;
      v8f g = vzero, u = vzero;
      const unsigned short* arow = xb + Mrow * HD;
      for (int k0 = 0; k0 < HD; k0 += 32) {
        const v16bf a = cat8(*(const v8bf*)(arow + k0 + 8 * khalf),
                             *(const v8bf*)(arow + k0 + 16 + 8 * khalf));
        const int kb = k0 + 16 * khalf;
        // prefetch next k-step's weight rows while WMMA pipeline drains
        __builtin_prefetch(gbase + (size_t)(kb + 32) * FD + ncol, 0, 0);
        __builtin_prefetch(ubase + (size_t)(kb + 32) * FD + ncol, 0, 0);
        v16bf bg, bu;
#pragma unroll
        for (int i = 0; i < 16; ++i) {
          const size_t off = (size_t)(k0 + (i < 8 ? 8 * khalf + i : 8 + 8 * khalf + i)) * FD + ncol;
          bg[i] = f2bf(gbase[off]);
          bu[i] = f2bf(ubase[off]);
        }
        g = __builtin_amdgcn_wmma_f32_16x16x32_bf16(false, a, false, bg,
                                                    (short)0, g, false, false);
        u = __builtin_amdgcn_wmma_f32_16x16x32_bf16(false, a, false, bu,
                                                    (short)0, u, false, false);
      }
#pragma unroll
      for (int r = 0; r < 8; ++r) {
        const int mr = r + 8 * khalf;         // C layout: M = r + 8*(lane>=16)
        const float gv = g[r];
        // silu via fast hw rcp (bf16-precision downstream anyway)
        const float mv = gv * __builtin_amdgcn_rcpf(1.f + __expf(-gv)) * u[r];
        mb[mr * FCHUNK + f0 + Mrow] = f2bfu(mv);
      }
    }
    __syncthreads();

    // Phase B: waves split H; accumulate m @ down over this chunk's K range
    {
      const unsigned short* arow = mb + Mrow * FCHUNK;
      for (int k0 = 0; k0 < FCHUNK; k0 += 32) {
        const v16bf a = cat8(*(const v8bf*)(arow + k0 + 8 * khalf),
                             *(const v8bf*)(arow + k0 + 16 + 8 * khalf));
        const int kb = fbase + k0 + 16 * khalf;
        __builtin_prefetch(dbase + (size_t)(kb + 32) * HD + wave * 128 + Mrow, 0, 0);
#pragma unroll
        for (int j = 0; j < 8; ++j) {
          const int ncol = wave * 128 + j * 16 + Mrow;
          v16bf b;
#pragma unroll
          for (int i = 0; i < 16; ++i)
            b[i] = f2bf(dbase[(size_t)(fbase + k0 + (i < 8 ? 8 * khalf + i : 8 + 8 * khalf + i)) * HD + ncol]);
          acc[j] = __builtin_amdgcn_wmma_f32_16x16x32_bf16(false, a, false, b,
                                                           (short)0, acc[j],
                                                           false, false);
        }
      }
    }
    __syncthreads();   // mb reused next chunk
  }

  // Epilogue: scale by routing weight, write deterministic slot partials
#pragma unroll
  for (int j = 0; j < 8; ++j) {
#pragma unroll
    for (int r = 0; r < 8; ++r) {
      const int mr = r + 8 * khalf;
      const float w = w_s[mr];
      if (w != 0.f) {
        const size_t o = ((size_t)slot_s[mr] * NT + tok_s[mr]) * HD
                       + wave * 128 + j * 16 + Mrow;
        part[o] = acc[j][r] * w;
      }
    }
  }
}

// ---------------- Combine the two routed slots per token -------------------
__global__ void moe_combine(const float* __restrict__ part,
                            float* __restrict__ out) {
  const int i = blockIdx.x * blockDim.x + threadIdx.x;
  if (i < NT * HD) {
    const float v = part[i] + part[(size_t)NT * HD + i];
    __builtin_nontemporal_store(v, out + i);   // out never re-read on device
  }
}

extern "C" void kernel_launch(void* const* d_in, const int* in_sizes, int n_in,
                              void* d_out, int out_size, void* d_ws, size_t ws_size,
                              hipStream_t stream) {
  const float* x  = (const float*)d_in[0];
  const float* rw = (const float*)d_in[1];
  const float* gp = (const float*)d_in[2];
  const float* up = (const float*)d_in[3];
  const float* dp = (const float*)d_in[4];
  float* out = (float*)d_out;

  char* ws = (char*)d_ws;
  float* wdense = (float*)ws;  ws += sizeof(float) * NT * NE;
  int*   cnt    = (int*)ws;    ws += sizeof(int) * 32;
  int*   toks   = (int*)ws;    ws += sizeof(int) * NE * NT;
  float* wts    = (float*)ws;  ws += sizeof(float) * NE * NT;
  int*   slots  = (int*)ws;    ws += sizeof(int) * NE * NT;
  float* part   = (float*)ws;  // 2 * NT * HD floats (~8 MB)

  moe_router <<<NT / 256, 256, 0, stream>>>(x, rw, wdense);
  moe_compact<<<1, 256, 0, stream>>>(wdense, cnt, toks, wts, slots);
  moe_ffn    <<<dim3(NT / TM, NE), 256, 0, stream>>>(x, gp, up, dp,
                                                     toks, wts, slots, cnt, part);
  moe_combine<<<(NT * HD) / 256, 256, 0, stream>>>(part, out);
}
